// TinyGPT_23562190585905
// MI455X (gfx1250) — matrix operand order repaired and checked by reading, not compiled, not run on hardware
//
#include <hip/hip_runtime.h>

// ---------------------------------------------------------------------------
// TinyGPT forward on MI455X (gfx1250, wave32, WMMA).
// - All GEMMs on v_wmma_f32_16x16x32_f16 (f16 in, f32 accumulate).
// - Weights converted+transposed to f16 [N,K] once per launch so both WMMA
//   operands are K-contiguous.
// - GEMM: 4 waves/block, block tile 16(M) x 256(N); the shared 16-row A panel
//   is staged into LDS with double-buffered global_load_async_to_lds_b128
//   (ASYNCcnt + s_wait_asynccnt), each wave runs 4 WMMA accumulator chains.
// - Attention: per-wave 16-query tile, scores in LDS, shuffle softmax, WMMA
//   for both QK^T and P.V.
// ---------------------------------------------------------------------------

typedef _Float16 h16;
typedef __attribute__((ext_vector_type(16))) _Float16 v16h;
typedef __attribute__((ext_vector_type(8)))  _Float16 v8h;
typedef __attribute__((ext_vector_type(8)))  float    v8f;

namespace {
constexpr int kB = 4;
constexpr int kT = 1024;
constexpr int kH = 12;
constexpr int kC = 768;
constexpr int kD = 64;
constexpr int kF = 3072;
constexpr int kV = 32000;
constexpr int kM = kB * kT;  // 4096 token rows
}

// ---------------------------------------------------------------------------
// x[b,t,:] = tok_emb[idx[b,t],:] + pos_emb[t,:]
// ---------------------------------------------------------------------------
__global__ void embed_kernel(const int* __restrict__ idx,
                             const float* __restrict__ tok,
                             const float* __restrict__ pos,
                             float* __restrict__ x) {
  int i = blockIdx.x * blockDim.x + threadIdx.x;
  if (i >= kM * kC) return;
  int row = i / kC;
  int c   = i - row * kC;
  int t   = row & (kT - 1);
  x[i] = tok[(size_t)idx[row] * kC + c] + pos[(size_t)t * kC + c];
}

// ---------------------------------------------------------------------------
// W [K,N] f32  ->  Wt [N,K] f16   (transpose so GEMM B-frags are K-contiguous)
// ---------------------------------------------------------------------------
__global__ void convtr_kernel(const float* __restrict__ W,
                              h16* __restrict__ Wt, int K, int N) {
  long i = (long)blockIdx.x * blockDim.x + threadIdx.x;
  if (i >= (long)K * N) return;
  int k = (int)(i / N);
  int n = (int)(i - (long)k * N);
  Wt[(size_t)n * K + k] = (h16)W[i];
}

// ---------------------------------------------------------------------------
// LayerNorm over C=768, one row per 256-thread block, f16 output.
// ---------------------------------------------------------------------------
__global__ __launch_bounds__(256)
void ln_kernel(const float* __restrict__ x,
               const float* __restrict__ gamma,
               const float* __restrict__ beta,
               h16* __restrict__ out) {
  const int row = blockIdx.x;
  const int tid = threadIdx.x;
  const float* xr = x + (size_t)row * kC;
  float v0 = xr[tid], v1 = xr[tid + 256], v2 = xr[tid + 512];
  float sum = v0 + v1 + v2;
  float sq  = v0 * v0 + v1 * v1 + v2 * v2;
  #pragma unroll
  for (int o = 16; o >= 1; o >>= 1) {
    sum += __shfl_xor(sum, o, 32);
    sq  += __shfl_xor(sq,  o, 32);
  }
  __shared__ float red[2][8];
  const int wid = tid >> 5, lane = tid & 31;
  if (lane == 0) { red[0][wid] = sum; red[1][wid] = sq; }
  __syncthreads();
  float ts = 0.f, tq = 0.f;
  #pragma unroll
  for (int i = 0; i < 8; ++i) { ts += red[0][i]; tq += red[1][i]; }
  const float mean = ts * (1.0f / kC);
  const float var  = tq * (1.0f / kC) - mean * mean;
  const float inv  = rsqrtf(var + 1e-5f);
  h16* orow = out + (size_t)row * kC;
  orow[tid]       = (h16)((v0 - mean) * inv * gamma[tid]       + beta[tid]);
  orow[tid + 256] = (h16)((v1 - mean) * inv * gamma[tid + 256] + beta[tid + 256]);
  orow[tid + 512] = (h16)((v2 - mean) * inv * gamma[tid + 512] + beta[tid + 512]);
}

// ---------------------------------------------------------------------------
// WMMA GEMM: D[M,N] = A[M,K](f16 row-major) * Bt[N,K](f16)^T (+bias,+epilogue)
// Block = 128 threads (4 waves); block tile 16 x 256, each wave owns 16 x 64
// (4 WMMA accumulator chains, A-frag reused 4x). The 16-row A panel is staged
// to LDS in 64-half K-chunks via async-to-LDS loads, double-buffered.
// MODE 0: f16 store                                    (Q, K projections)
// MODE 1: f16 store transposed to vt[(b*H+h)*D+d][t]   (V projection)
// MODE 2: f32 residual  Res += acc + bias              (attn proj / MLP down)
// MODE 3: f16 store of exact GELU(acc + bias)          (MLP up)
// MODE 4: f32 store                                    (logits)
// ---------------------------------------------------------------------------
template <int MODE>
__global__ __launch_bounds__(128)
void gemm_kernel(const h16* __restrict__ A, const h16* __restrict__ Bt,
                 const float* __restrict__ bias, void* __restrict__ Out,
                 float* __restrict__ Res, int M, int N, int K) {
  __shared__ __align__(16) h16 abuf[2][16 * 64];  // double-buffered A panel

  const int wave = threadIdx.x >> 5;
  const int lane = threadIdx.x & 31;
  const int m0 = blockIdx.x * 16;
  const int n0 = (blockIdx.y * 4 + wave) * 64;

  const int mrow = lane & 15;  // tile row (A) / tile col (B,D) for this lane
  const int hi   = lane >> 4;  // lane-half selects K = hi*16 + [0..15] (contiguous order)

  // Async A-panel staging: each thread moves one 16B segment per chunk.
  const int srow = threadIdx.x >> 3;       // 0..15
  const int sseg = threadIdx.x & 7;        // 8 x 16B = 128B = 64 halfs per row
  const unsigned lds_base = (unsigned)(size_t)&abuf[0][0];
  const unsigned lds_slot = lds_base + (unsigned)(srow * 128 + sseg * 16);
  auto issue_chunk = [&](int c, int buf) {
    unsigned lds = lds_slot + (unsigned)(buf * 2048);
    unsigned gof = (unsigned)(((size_t)(m0 + srow) * K + c * 64 + sseg * 8) * 2);
    asm volatile("global_load_async_to_lds_b128 %0, %1, %2"
                 :: "v"(lds), "v"(gof), "s"(A) : "memory");
  };

  // Per-wave B row base pointers (one per 16-wide N subtile).
  const h16* brow[4];
  #pragma unroll
  for (int ns = 0; ns < 4; ++ns)
    brow[ns] = Bt + (size_t)(n0 + ns * 16 + mrow) * K + hi * 16;

  v8f acc[4] = {v8f{}, v8f{}, v8f{}, v8f{}};
  const int nch = K / 64;

  issue_chunk(0, 0);
  for (int c = 0; c < nch; ++c) {
    asm volatile("s_wait_asynccnt 0" ::: "memory");
    __syncthreads();
    if (c + 1 < nch) issue_chunk(c + 1, (c + 1) & 1);

    const h16* apanel = &abuf[c & 1][0];
    #pragma unroll
    for (int k0 = 0; k0 < 64; k0 += 32) {
      v16h a;
      // A frag (16x32 f16) from LDS: lane-half hi holds K = hi*16 + [0..15]
      // within this 32-wide K step, contiguous.
      const h16* ap = apanel + mrow * 64 + k0 + hi * 16;
      *(v8h*)&a       = *(const v8h*)ap;
      *((v8h*)&a + 1) = *(const v8h*)(ap + 8);
      #pragma unroll
      for (int ns = 0; ns < 4; ++ns) {
        // B frag (32x16 f16): lane n holds K = hi*16 + [0..15], contiguous.
        v16h b = *(const v16h*)(brow[ns] + c * 64 + k0);
        acc[ns] = __builtin_amdgcn_wmma_f32_16x16x32_f16(
            false, a, false, b, (short)0, acc[ns], false, false);
      }
    }
  }

  #pragma unroll
  for (int ns = 0; ns < 4; ++ns) {
    const int n = n0 + ns * 16 + mrow;
    const float bv = bias ? bias[n] : 0.0f;
    #pragma unroll
    for (int r = 0; r < 8; ++r) {
      const int m = m0 + r + hi * 8;  // C/D layout: lane-half -> M+8
      const size_t oi = (size_t)m * N + n;
      const float val = acc[ns][r] + bv;
      if constexpr (MODE == 0) {
        ((h16*)Out)[oi] = (h16)val;
      } else if constexpr (MODE == 1) {
        const int bb = m >> 10, tt = m & (kT - 1);
        const int head = n >> 6, d = n & (kD - 1);
        ((h16*)Out)[((size_t)(bb * kH + head) * kD + d) * kT + tt] = (h16)val;
      } else if constexpr (MODE == 2) {
        Res[oi] += val;
      } else if constexpr (MODE == 3) {
        const float g = 0.5f * val * (1.0f + erff(val * 0.70710678118f));
        ((h16*)Out)[oi] = (h16)g;
      } else {
        ((float*)Out)[oi] = val;
      }
    }
  }
}

// ---------------------------------------------------------------------------
// Causal attention: one wave per (b, head, 16-query tile).
// S = Q.K^T via WMMA -> LDS f32 [16][1024]; masked shuffle-softmax;
// P (f16, LDS) reread in A-frag layout; y = P.V via 4 WMMA chains (D=64).
// vt is V pre-transposed to [(b*H+h)*D + d][t] so B-frags are key-contiguous.
// ---------------------------------------------------------------------------
__global__ __launch_bounds__(32)
void attn_kernel(const h16* __restrict__ q, const h16* __restrict__ k,
                 const h16* __restrict__ vt, h16* __restrict__ y) {
  extern __shared__ char smem[];
  float* sc = (float*)smem;                        // [16][1024] scores
  h16*   pb = (h16*)(smem + 16 * 1024 * 4);        // [16][1024] probs f16

  const int lane = threadIdx.x;
  const int qt   = blockIdx.x;   // query tile (T/16 = 64)
  const int head = blockIdx.y;   // 12
  const int b    = blockIdx.z;   // 4
  const int nkt  = qt + 1;       // causal: key tiles 0..qt
  const int nkeys = nkt * 16;

  const int mrow = lane & 15;
  const int hi   = lane >> 4;

  // Q fragments for K-dim 64 (two 16x32 chunks)
  const h16* qrow = q + (size_t)(b * kT + qt * 16 + mrow) * kC + head * kD;
  v16h qa[2];
  #pragma unroll
  for (int kc = 0; kc < 2; ++kc) {
    *(v8h*)&qa[kc]       = *(const v8h*)(qrow + kc * 32 + hi * 16);
    *((v8h*)&qa[kc] + 1) = *(const v8h*)(qrow + kc * 32 + 8 + hi * 16);
  }

  const float scale = 0.125f;  // 1/sqrt(64)
  for (int kt = 0; kt < nkt; ++kt) {
    const h16* krow =
        k + (size_t)(b * kT + kt * 16 + mrow) * kC + head * kD + hi * 16;
    v8f s = {};
    #pragma unroll
    for (int kc = 0; kc < 2; ++kc) {
      v16h bf = *(const v16h*)(krow + kc * 32);
      s = __builtin_amdgcn_wmma_f32_16x16x32_f16(
          false, qa[kc], false, bf, (short)0, s, false, false);
    }
    const int keyg = kt * 16 + mrow;          // lane's key column
    #pragma unroll
    for (int r = 0; r < 8; ++r) {
      const int m = r + hi * 8;               // query row within tile
      const bool masked = (kt == qt) && (keyg > qt * 16 + m);
      sc[m * 1024 + kt * 16 + mrow] = masked ? -1e30f : s[r] * scale;
    }
  }

  // Row softmax (wave32 lockstep; DS ops same-wave in order)
  for (int m = 0; m < 16; ++m) {
    float mx = -3e38f;
    for (int c = lane; c < nkeys; c += 32) mx = fmaxf(mx, sc[m * 1024 + c]);
    #pragma unroll
    for (int o = 16; o >= 1; o >>= 1) mx = fmaxf(mx, __shfl_xor(mx, o, 32));
    float sum = 0.f;
    for (int c = lane; c < nkeys; c += 32) {
      float e = __expf(sc[m * 1024 + c] - mx);
      sc[m * 1024 + c] = e;
      sum += e;
    }
    #pragma unroll
    for (int o = 16; o >= 1; o >>= 1) sum += __shfl_xor(sum, o, 32);
    const float inv = 1.0f / sum;
    for (int c = lane; c < nkeys; c += 32)
      pb[m * 1024 + c] = (h16)(sc[m * 1024 + c] * inv);
  }
  // zero-pad probs to a multiple of 32 keys for the last WMMA chunk
  if (nkt & 1) {
    for (int i = lane; i < 256; i += 32)
      pb[(i >> 4) * 1024 + nkeys + (i & 15)] = (h16)0.f;
  }

  // y = P.V : 4 WMMA accumulator chains over D=64
  const int nchunks = (nkeys + 31) / 32;
  v8f yacc[4] = {v8f{}, v8f{}, v8f{}, v8f{}};
  for (int ch = 0; ch < nchunks; ++ch) {
    const int kb = ch * 32;
    v16h pa;
    const h16* prow = pb + mrow * 1024 + kb;
    *(v8h*)&pa       = *(const v8h*)(prow + hi * 16);
    *((v8h*)&pa + 1) = *(const v8h*)(prow + 8 + hi * 16);
    #pragma unroll
    for (int nc = 0; nc < 4; ++nc) {
      const h16* vrow =
          vt + (size_t)((b * kH + head) * kD + nc * 16 + mrow) * kT + kb + hi * 16;
      v16h vb = *(const v16h*)vrow;
      yacc[nc] = __builtin_amdgcn_wmma_f32_16x16x32_f16(
          false, pa, false, vb, (short)0, yacc[nc], false, false);
    }
  }

  #pragma unroll
  for (int nc = 0; nc < 4; ++nc) {
    #pragma unroll
    for (int r = 0; r < 8; ++r) {
      const int m = qt * 16 + r + hi * 8;
      const int d = nc * 16 + mrow;
      y[(size_t)(b * kT + m) * kC + head * kD + d] = (h16)yacc[nc][r];
    }
  }
}

// ---------------------------------------------------------------------------
// Host orchestration
// ---------------------------------------------------------------------------
static inline size_t alignup(size_t x) { return (x + 255) & ~(size_t)255; }

extern "C" void kernel_launch(void* const* d_in, const int* in_sizes, int n_in,
                              void* d_out, int out_size, void* d_ws, size_t ws_size,
                              hipStream_t stream) {
  (void)in_sizes; (void)n_in; (void)out_size; (void)ws_size;
  const int*   idx   = (const int*)  d_in[0];
  const float* tok   = (const float*)d_in[1];
  const float* pos   = (const float*)d_in[2];
  const float* Wq    = (const float*)d_in[3];
  const float* Wk    = (const float*)d_in[4];
  const float* Wv    = (const float*)d_in[5];
  const float* Wp    = (const float*)d_in[6];
  const float* bp    = (const float*)d_in[7];
  const float* ln1s  = (const float*)d_in[8];
  const float* ln1b  = (const float*)d_in[9];
  const float* ln2s  = (const float*)d_in[10];
  const float* ln2b  = (const float*)d_in[11];
  const float* W1    = (const float*)d_in[12];
  const float* b1    = (const float*)d_in[13];
  const float* W2    = (const float*)d_in[14];
  const float* b2    = (const float*)d_in[15];
  const float* lnfs  = (const float*)d_in[16];
  const float* lnfb  = (const float*)d_in[17];
  const float* Whead = (const float*)d_in[18];
  float* out = (float*)d_out;

  char* p = (char*)d_ws;
  auto carve = [&](size_t bytes) { char* r = p; p += alignup(bytes); return r; };
  float* x   = (float*)carve((size_t)kM * kC * 4);
  h16* h     = (h16*)carve((size_t)kM * kC * 2);
  h16* qb    = (h16*)carve((size_t)kM * kC * 2);
  h16* kb    = (h16*)carve((size_t)kM * kC * 2);
  h16* vtb   = (h16*)carve((size_t)kM * kC * 2);
  h16* yb    = (h16*)carve((size_t)kM * kC * 2);
  h16* ub    = (h16*)carve((size_t)kM * kF * 2);
  h16* wqT   = (h16*)carve((size_t)kC * kC * 2);
  h16* wkT   = (h16*)carve((size_t)kC * kC * 2);
  h16* wvT   = (h16*)carve((size_t)kC * kC * 2);
  h16* wpT   = (h16*)carve((size_t)kC * kC * 2);
  h16* w1T   = (h16*)carve((size_t)kC * kF * 2);
  h16* w2T   = (h16*)carve((size_t)kF * kC * 2);
  h16* whT   = (h16*)carve((size_t)kC * kV * 2);

  const dim3 b256(256);
  const dim3 gCC(((size_t)kC * kC + 255) / 256);
  const dim3 gCF(((size_t)kC * kF + 255) / 256);

  convtr_kernel<<<dim3(((size_t)kC * kV + 255) / 256), b256, 0, stream>>>(Whead, whT, kC, kV);
  embed_kernel<<<dim3((kM * kC + 255) / 256), b256, 0, stream>>>(idx, tok, pos, x);

  for (int l = 0; l < 4; ++l) {
    convtr_kernel<<<gCC, b256, 0, stream>>>(Wq + (size_t)l * kC * kC, wqT, kC, kC);
    convtr_kernel<<<gCC, b256, 0, stream>>>(Wk + (size_t)l * kC * kC, wkT, kC, kC);
    convtr_kernel<<<gCC, b256, 0, stream>>>(Wv + (size_t)l * kC * kC, wvT, kC, kC);
    convtr_kernel<<<gCC, b256, 0, stream>>>(Wp + (size_t)l * kC * kC, wpT, kC, kC);
    convtr_kernel<<<gCF, b256, 0, stream>>>(W1 + (size_t)l * kC * kF, w1T, kC, kF);
    convtr_kernel<<<gCF, b256, 0, stream>>>(W2 + (size_t)l * kF * kC, w2T, kF, kC);

    ln_kernel<<<dim3(kM), b256, 0, stream>>>(x, ln1s + l * kC, ln1b + l * kC, h);
    gemm_kernel<0><<<dim3(kM / 16, kC / 256), dim3(128), 0, stream>>>(
        h, wqT, nullptr, qb, nullptr, kM, kC, kC);
    gemm_kernel<0><<<dim3(kM / 16, kC / 256), dim3(128), 0, stream>>>(
        h, wkT, nullptr, kb, nullptr, kM, kC, kC);
    gemm_kernel<1><<<dim3(kM / 16, kC / 256), dim3(128), 0, stream>>>(
        h, wvT, nullptr, vtb, nullptr, kM, kC, kC);
    attn_kernel<<<dim3(kT / 16, kH, kB), dim3(32), 96 * 1024, stream>>>(qb, kb, vtb, yb);
    gemm_kernel<2><<<dim3(kM / 16, kC / 256), dim3(128), 0, stream>>>(
        yb, wpT, bp + l * kC, nullptr, x, kM, kC, kC);

    ln_kernel<<<dim3(kM), b256, 0, stream>>>(x, ln2s + l * kC, ln2b + l * kC, h);
    gemm_kernel<3><<<dim3(kM / 16, kF / 256), dim3(128), 0, stream>>>(
        h, w1T, b1 + l * kF, ub, nullptr, kM, kF, kC);
    gemm_kernel<2><<<dim3(kM / 16, kC / 256), dim3(128), 0, stream>>>(
        ub, w2T, b2 + l * kC, nullptr, x, kM, kC, kF);
  }

  ln_kernel<<<dim3(kM), b256, 0, stream>>>(x, lnfs, lnfb, h);
  gemm_kernel<4><<<dim3(kM / 16, kV / 256), dim3(128), 0, stream>>>(
      h, whT, nullptr, out, nullptr, kM, kV, kC);
}
